// CRF_21474836480621
// MI455X (gfx1250) — matrix operand order repaired and checked by reading, not compiled, not run on hardware
//
#include <hip/hip_runtime.h>
#include <hip/hip_bf16.h>
#include <stdint.h>

typedef __attribute__((ext_vector_type(16))) _Float16 v16h;
typedef __attribute__((ext_vector_type(8)))  float    v8f;

#define SEQ   512
#define NBAT  512
#define NT    64
#define BTILE 16   // batch rows per block
#define NTHR  128  // 4 waves of 32

// ---------------- async emission tile copy (16 rows x 64 f32 = 4KB, contiguous) ----
__device__ __forceinline__ void async_copy_em(const float* __restrict__ src,
                                              float* dstLDS, int tid) {
    // 4096 bytes = 256 x b128 chunks; 128 threads x 2 chunks each.
    uint64_t ga = (uint64_t)(uintptr_t)src + (uint32_t)tid * 16u;
    uint32_t la = (uint32_t)(uintptr_t)dstLDS + (uint32_t)tid * 16u;
    asm volatile("global_load_async_to_lds_b128 %0, %1, off" :: "v"(la), "v"(ga) : "memory");
    ga += 2048u; la += 2048u;
    asm volatile("global_load_async_to_lds_b128 %0, %1, off" :: "v"(la), "v"(ga) : "memory");
}
__device__ __forceinline__ void wait_async0() {
    asm volatile("s_wait_asynccnt 0" ::: "memory");
}

// ---------------- kernel 1: numerator (gather score) -> ws[b] ---------------------
__global__ __launch_bounds__(256) void crf_score_k(
    const float* __restrict__ em, const int* __restrict__ tags,
    const int* __restrict__ mask, const float* __restrict__ startT,
    const float* __restrict__ endT, const float* __restrict__ trans,
    float* __restrict__ ws)
{
    int b = blockIdx.x * blockDim.x + threadIdx.x;
    if (b >= NBAT) return;
    int tg0 = tags[b];
    float score = startT[tg0] + em[(size_t)b * NT + tg0];
    int prev = tg0, last = tg0;
    for (int t = 1; t < SEQ; ++t) {
        int   tg = tags[t * NBAT + b];
        int   mk = mask[t * NBAT + b];
        float e  = em[((size_t)t * NBAT + b) * NT + tg];
        score += (trans[prev * NT + tg] + e) * (float)mk;
        if (mk) last = tg;
        prev = tg;
    }
    score += endT[last];
    ws[b] = score;
}

// ---------------- kernel 2: normalizer (scaled forward algorithm via WMMA) --------
// alpha'[m,j] = exp(em[t,m,j]) * sum_i alpha[m,i] * exp(T[i,j]);  renormalize rows,
// accumulate log-scale.  One block = 16 batch rows; wave w owns j-tile [16w,16w+16).
__global__ __launch_bounds__(NTHR) void crf_norm_k(
    const float* __restrict__ em, const int* __restrict__ mask,
    const float* __restrict__ startT, const float* __restrict__ endT,
    const float* __restrict__ trans, float* __restrict__ ws_norm)
{
    __shared__ _Float16 s_expT[NT * NT];                 // exp(transitions), 8 KB
    __shared__ _Float16 s_alpha[BTILE * NT];             // normalized alpha, 2 KB
    __shared__ __align__(16) float s_em[2][BTILE * NT];  // double-buffered emissions, 8 KB
    __shared__ float s_v[BTILE * NT];                    // unnormalized next alpha, 4 KB
    __shared__ float s_pm[BTILE * 4];                    // partial row maxima
    __shared__ float s_inv[BTILE];
    __shared__ float s_L[BTILE];                         // per-row log scale
    __shared__ int   s_mk[BTILE];

    const int tid  = threadIdx.x;
    const int lane = tid & 31;
    const int w    = tid >> 5;     // wave id 0..3 -> j-tile
    const int g    = lane >> 4;    // half-wave
    const int nn   = lane & 15;
    const int b0   = blockIdx.x * BTILE;

    // exp(T) into LDS (f16) -- constant B matrix for the whole scan
    for (int i = tid; i < NT * NT; i += NTHR)
        s_expT[i] = (_Float16)__expf(trans[i]);

    // async-load emissions for t=0
    async_copy_em(em + (size_t)b0 * NT, s_em[0], tid);
    wait_async0();
    __syncthreads();

    // B fragments in registers: B[k, n] with n = 16w + nn, lane-halves cover K by 16s
    v16h bf0, bf1;
    {
        const int n = w * 16 + nn;
        #pragma unroll
        for (int e = 0; e < 16; ++e) {
            bf0[e] = s_expT[(g * 16 + e) * NT + n];          // K = 0..31
            bf1[e] = s_expT[(32 + g * 16 + e) * NT + n];     // K = 32..63
        }
    }

    // init: score0 = startT + em[0];  alpha0 = exp(score0 - rowmax), L = rowmax
    if (tid < BTILE) {
        float mx = -3.0e38f;
        for (int j = 0; j < NT; ++j)
            mx = fmaxf(mx, startT[j] + s_em[0][tid * NT + j]);
        s_L[tid] = mx;
    }
    __syncthreads();
    for (int i = tid; i < BTILE * NT; i += NTHR) {
        int m = i >> 6, j = i & (NT - 1);
        s_alpha[i] = (_Float16)__expf(startT[j] + s_em[0][i] - s_L[m]);
    }
    // prefetch em[1]
    async_copy_em(em + ((size_t)1 * NBAT + b0) * NT, s_em[1], tid);
    __syncthreads();

    for (int t = 1; t < SEQ; ++t) {
        float* emb = s_em[t & 1];
        wait_async0();
        __syncthreads();
        if (t + 1 < SEQ)
            async_copy_em(em + ((size_t)(t + 1) * NBAT + b0) * NT, s_em[(t + 1) & 1], tid);
        if (tid >= NTHR - BTILE)
            s_mk[tid - (NTHR - BTILE)] = mask[t * NBAT + b0 + (tid - (NTHR - BTILE))];

        // A fragments from current alpha: lane holds row m = nn, lane-halves cover K by 16s
        v16h af0, af1;
        #pragma unroll
        for (int e = 0; e < 16; ++e) {
            int k = g * 16 + e;                              // K = 0..31
            af0[e] = s_alpha[nn * NT + k];
            af1[e] = s_alpha[nn * NT + 32 + k];              // K = 32..63
        }

        v8f acc = {};
        acc = __builtin_amdgcn_wmma_f32_16x16x32_f16(false, af0, false, bf0,
                                                     (short)0, acc, false, false);
        acc = __builtin_amdgcn_wmma_f32_16x16x32_f16(false, af1, false, bf1,
                                                     (short)0, acc, false, false);

        // v[m,j] = acc * exp(em[t,m,j]);  D layout: vgpr r -> m = r + 8*g, col n = nn
        #pragma unroll
        for (int r = 0; r < 8; ++r) {
            int mm = r + 8 * g;
            int j  = w * 16 + nn;
            s_v[mm * NT + j] = acc[r] * __expf(emb[mm * NT + j]);
        }
        __syncthreads();

        // cross-wave row max (64 j values per row)
        if (tid < 64) {
            int mm = tid & 15, q = tid >> 4;
            float mx = s_v[mm * NT + q * 16];
            #pragma unroll
            for (int j = 1; j < 16; ++j)
                mx = fmaxf(mx, s_v[mm * NT + q * 16 + j]);
            s_pm[mm * 4 + q] = mx;
        }
        __syncthreads();
        if (tid < BTILE) {
            if (s_mk[tid]) {
                float r4 = fmaxf(fmaxf(s_pm[tid * 4], s_pm[tid * 4 + 1]),
                                 fmaxf(s_pm[tid * 4 + 2], s_pm[tid * 4 + 3]));
                s_inv[tid] = 1.0f / r4;
                s_L[tid]  += __logf(r4);
            }
        }
        __syncthreads();
        for (int i = tid; i < BTILE * NT; i += NTHR) {
            int mm = i >> 6;
            if (s_mk[mm]) s_alpha[i] = (_Float16)(s_v[i] * s_inv[mm]);
        }
        __syncthreads();
    }

    // normalizer[m] = L[m] + log( sum_j alpha[m,j] * exp(endT[j]) )
    if (tid < BTILE) {
        float s = 0.0f;
        for (int j = 0; j < NT; ++j)
            s += (float)s_alpha[tid * NT + j] * __expf(endT[j]);
        ws_norm[b0 + tid] = s_L[tid] + __logf(s);
    }
}

// ---------------- kernel 3: deterministic final reduction -------------------------
__global__ __launch_bounds__(256) void crf_reduce_k(const float* __restrict__ ws,
                                                    float* __restrict__ out)
{
    __shared__ float sm[256];
    int tid = threadIdx.x;
    // llh[b] = score[b] - norm[b]; fixed summation order -> deterministic
    float s = (ws[tid] - ws[NBAT + tid]) + (ws[tid + 256] - ws[NBAT + tid + 256]);
    sm[tid] = s;
    __syncthreads();
    #pragma unroll
    for (int off = 128; off > 0; off >>= 1) {
        if (tid < off) sm[tid] += sm[tid + off];
        __syncthreads();
    }
    if (tid == 0) out[0] = sm[0];
}

// ---------------- launch ----------------------------------------------------------
extern "C" void kernel_launch(void* const* d_in, const int* in_sizes, int n_in,
                              void* d_out, int out_size, void* d_ws, size_t ws_size,
                              hipStream_t stream) {
    const float* em     = (const float*)d_in[0];
    const int*   tags   = (const int*)  d_in[1];
    const int*   mask   = (const int*)  d_in[2];
    const float* startT = (const float*)d_in[3];
    const float* endT   = (const float*)d_in[4];
    const float* trans  = (const float*)d_in[5];
    float* ws  = (float*)d_ws;           // [0..511] scores, [512..1023] normalizers
    float* out = (float*)d_out;

    crf_score_k<<<(NBAT + 255) / 256, 256, 0, stream>>>(em, tags, mask, startT, endT, trans, ws);
    crf_norm_k<<<NBAT / BTILE, NTHR, 0, stream>>>(em, mask, startT, endT, trans, ws + NBAT);
    crf_reduce_k<<<1, 256, 0, stream>>>(ws, out);
}
